// Chambers_15590731285000
// MI455X (gfx1250) — matrix operand order repaired and checked by reading, not compiled, not run on hardware
//
#include <hip/hip_runtime.h>
#include <hip/hip_bf16.h>

typedef __attribute__((ext_vector_type(16))) _Float16 v16h;
typedef __attribute__((ext_vector_type(8)))  _Float16 v8h;
typedef __attribute__((ext_vector_type(8)))  float    v8f;

#define N_B 262144
#define RES_DIM 100

// ---- WMMA fragment loaders (contiguous k order for A and B, wave32) ----

// A-matrix (16-bit, 16x32): lane m = lane&15; lanes 0-15 hold K 0-15,
// lanes 16-31 hold K 16-31 (elements 0-15 = k0..k0+15), matching B.
__device__ inline v16h ld_afrag(const _Float16* p0) {
    v8h lo = *reinterpret_cast<const v8h*>(p0);
    v8h hi = *reinterpret_cast<const v8h*>(p0 + 8);
    v16h r;
#pragma unroll
    for (int i = 0; i < 8; ++i) { r[i] = lo[i]; r[i + 8] = hi[i]; }
    return r;
}

// B-matrix (16-bit, 32x16): lane n = lane&15; lanes 0-15 hold K 0-15,
// lanes 16-31 hold K 16-31, consecutive halves. B[k][n] = W[n][k] -> one
// contiguous 32-byte chunk of weight row n.
__device__ inline v16h ld_bfrag(const _Float16* p0) {
    return *reinterpret_cast<const v16h*>(p0);
}

__device__ inline float silu_f(float x) {
    return x / (1.0f + __expf(-x));
}

// One MLP layer on a 32-row tile (two 16-row M-tiles per wave so each B
// fragment feeds two WMMAs -> half the LDS B-traffic per MAC).
// Out[32][NT*16] = silu(A[32][KT*32] * W^T + b). All 32 lanes active.
template <int KT, int NT>
__device__ inline void mlp_layer2(const _Float16* __restrict__ A, int astride,
                                  const _Float16* __restrict__ W, int wstride,
                                  const float* __restrict__ bias,
                                  _Float16* __restrict__ Out, int ostride,
                                  int lane) {
    const int mrow = lane & 15;
    const int khi  = lane >> 4;

    v16h a0[KT], a1[KT];
#pragma unroll
    for (int kt = 0; kt < KT; ++kt) {
        a0[kt] = ld_afrag(A + mrow * astride        + kt * 32 + khi * 16);
        a1[kt] = ld_afrag(A + (16 + mrow) * astride + kt * 32 + khi * 16);
    }

#pragma unroll
    for (int nt = 0; nt < NT; ++nt) {
        v8f acc0 = {};
        v8f acc1 = {};
#pragma unroll
        for (int kt = 0; kt < KT; ++kt) {
            v16h bfrag = ld_bfrag(W + (nt * 16 + mrow) * wstride + kt * 32 + khi * 16);
            acc0 = __builtin_amdgcn_wmma_f32_16x16x32_f16(
                false, a0[kt], false, bfrag, (short)0, acc0, false, false);
            acc1 = __builtin_amdgcn_wmma_f32_16x16x32_f16(
                false, a1[kt], false, bfrag, (short)0, acc1, false, false);
        }
        const int   n  = nt * 16 + mrow;
        const float bv = bias[n];
#pragma unroll
        for (int v = 0; v < 8; ++v) {
            const int m = khi * 8 + v;               // C/D layout: m = v + 8*(lane>=16)
            Out[m * ostride + n]        = (_Float16)silu_f(acc0[v] + bv);
            Out[(16 + m) * ostride + n] = (_Float16)silu_f(acc1[v] + bv);
        }
    }
}

__global__ __launch_bounds__(256) void Chambers_kernel(
    const float* __restrict__ res,
    const float* __restrict__ W1, const float* __restrict__ b1,
    const float* __restrict__ W2, const float* __restrict__ b2,
    const float* __restrict__ W3, const float* __restrict__ b3,
    const float* __restrict__ W4, const float* __restrict__ b4,
    const float* __restrict__ coupling,
    float* __restrict__ out) {
    extern __shared__ char smem[];
    // LDS layout (bytes):
    //   [0,      32768)  W1 f16 [128][128]  (K padded 100->128)
    //   [32768,  49152)  W2 f16 [64][128]
    //   [49152,  53248)  W3 f16 [32][64]
    //   [53248, 118784)  res tiles, 8 waves x [32][128] f16 (8192 B each)
    //   [118784,184320)  h1 tiles,  8 waves x [32][128] f16
    //   [184320,217088)  h2 tiles,  8 waves x [32][64]  f16 (4096 B each)
    //   [217088,233472)  h3 tiles,  8 waves x [32][32]  f16 (2048 B each)
    //   [233472,239616)  raw,       8 waves x [32][6]   f32 (768 B each)
    _Float16* w1T = (_Float16*)(smem);
    _Float16* w2T = (_Float16*)(smem + 32768);
    _Float16* w3T = (_Float16*)(smem + 49152);

    const int tid  = threadIdx.x;
    const int wave = tid >> 5;
    const int lane = tid & 31;

    _Float16* resT = (_Float16*)(smem + 53248  + wave * 8192);
    _Float16* h1T  = (_Float16*)(smem + 118784 + wave * 8192);
    _Float16* h2T  = (_Float16*)(smem + 184320 + wave * 4096);
    _Float16* h3T  = (_Float16*)(smem + 217088 + wave * 2048);
    float*    rawB = (float*)   (smem + 233472 + wave * 768);

    const long long row0 = (long long)blockIdx.x * 256 + wave * 32;

    // Stage this wave's 32 residual rows: f32 -> f16, zero-pad K to 128.
    for (int i = lane; i < 32 * 128; i += 32) {
        const int r = i >> 7, k = i & 127;
        const float v = (k < RES_DIM) ? res[(row0 + r) * RES_DIM + k] : 0.0f;
        resT[i] = (_Float16)v;
    }

    for (int c = 0; c < 6; ++c) {
        __syncthreads();  // all waves done reading previous chamber's weights
        // Stage chamber-c weights (whole block cooperatively), f32 -> f16.
        for (int i = tid; i < 128 * 128; i += 256) {
            const int o = i >> 7, k = i & 127;
            w1T[i] = (k < RES_DIM) ? (_Float16)W1[(c * 128 + o) * RES_DIM + k]
                                   : (_Float16)0.0f;
        }
        for (int i = tid; i < 64 * 128; i += 256)
            w2T[i] = (_Float16)W2[c * 64 * 128 + i];
        for (int i = tid; i < 32 * 64; i += 256)
            w3T[i] = (_Float16)W3[c * 32 * 64 + i];
        __syncthreads();

        // Three WMMA layers on this wave's 32-row tile.
        mlp_layer2<4, 8>(resT, 128, w1T, 128, b1 + c * 128, h1T, 128, lane);
        mlp_layer2<4, 4>(h1T, 128, w2T, 128, b2 + c * 64,  h2T, 64,  lane);
        mlp_layer2<2, 2>(h2T, 64,  w3T, 64,  b3 + c * 32,  h3T, 32,  lane);

        // Layer 4: [32][32] x [32] -> [32]; one row per lane, VALU dot.
        {
            float acc = b4[c];
#pragma unroll
            for (int p = 0; p < 32; ++p)
                acc += (float)h3T[lane * 32 + p] * W4[c * 32 + p];
            rawB[lane * 6 + c] = acc;
        }
    }

    // Coupling-flow epilogue: every lane owns one of this wave's 32 rows.
    {
        float a[6], rw[6];
#pragma unroll
        for (int c = 0; c < 6; ++c) {
            rw[c] = rawB[lane * 6 + c];
            a[c]  = 1.0f / (1.0f + __expf(-rw[c]));
        }
        const float decay[6] = {0.9f, 0.93f, 0.85f, 0.97f, 0.88f, 0.94f};
#pragma unroll
        for (int it = 0; it < 5; ++it) {
            float na[6];
#pragma unroll
            for (int i = 0; i < 6; ++i) a[i] *= decay[i];
#pragma unroll
            for (int i = 0; i < 6; ++i) {
                float s = 0.0f;
#pragma unroll
                for (int j = 0; j < 6; ++j)
                    s += coupling[i * 6 + j] * 0.02f * __sinf(a[j] - a[i]);
                na[i] = a[i] + s;
            }
#pragma unroll
            for (int i = 0; i < 6; ++i)
                a[i] = fminf(fmaxf(na[i], 0.0f), 1.0f);
        }
        const long long row = row0 + lane;
#pragma unroll
        for (int c = 0; c < 6; ++c) {
            out[row * 6 + c] = a[c];                               // act
            out[(long long)N_B * 6 + row * 6 + c] = rw[c];         // raw
        }
    }
}

extern "C" void kernel_launch(void* const* d_in, const int* in_sizes, int n_in,
                              void* d_out, int out_size, void* d_ws, size_t ws_size,
                              hipStream_t stream) {
    const float* res      = (const float*)d_in[0];
    const float* W1       = (const float*)d_in[1];
    const float* b1       = (const float*)d_in[2];
    const float* W2       = (const float*)d_in[3];
    const float* b2       = (const float*)d_in[4];
    const float* W3       = (const float*)d_in[5];
    const float* b3       = (const float*)d_in[6];
    const float* W4       = (const float*)d_in[7];
    const float* b4       = (const float*)d_in[8];
    const float* coupling = (const float*)d_in[9];
    float* out = (float*)d_out;

    const int rows_per_block = 256;                 // 8 waves x 32 rows
    const int grid = N_B / rows_per_block;          // 1024
    const size_t smem_bytes = 239616;               // ~234 KB LDS per workgroup

    Chambers_kernel<<<grid, 256, smem_bytes, stream>>>(
        res, W1, b1, W2, b2, W3, b3, W4, b4, coupling, out);
}
